// GNN_62259845923325
// MI455X (gfx1250) — matrix operand order repaired and checked by reading, not compiled, not run on hardware
//
#include <hip/hip_runtime.h>
#include <hip/hip_bf16.h>
#include <math.h>

// ---------------------------------------------------------------------------
// Types for CDNA5 WMMA (wave32, 16x16x32 bf16 -> f32 accumulate)
// ---------------------------------------------------------------------------
typedef __attribute__((ext_vector_type(16))) __bf16 v16bf;
typedef __attribute__((ext_vector_type(8)))  float  v8f;

#define NEG_SLOPE 0.2f

// ---------------------------------------------------------------------------
// GEMM: Hout[nrows x 64] = X[nrows x K] * W[K x 64], fp32 in/out, bf16 WMMA.
//  - W is staged once per workgroup into LDS, converted to bf16 and swizzled
//    into fragment order: [kstep][half][ctile][lane15][j] so a B fragment is
//    one aligned 32-byte LDS read per lane.
//  - A rows are read as float4 (one contiguous 16-float chunk per lane per
//    K-step: k = 16*half + [0..16), same k order as the B fragments) and packed to bf16 with cvt_pk.
//  - One wave computes a 16-row x 64-col stripe (4 col tiles, A reused).
// Requires: nrows % 16 == 0, K % 32 == 0, K <= 128, ncols == 64.
// ---------------------------------------------------------------------------
__global__ __launch_bounds__(256)
void gat_gemm_bf16_wmma(const float* __restrict__ X,
                        const float* __restrict__ W,
                        float* __restrict__ Hout,
                        int nrows, int K) {
  const int NCOLS = 64;
  __shared__ __align__(32) __bf16 ldsW[128 * 64];   // 16 KB max (K=128)

  // ---- stage W -> LDS (bf16, fragment-swizzled), all 256 threads ----------
  // element (k, n): kstep s = k/32, half h = (k%32)/16, j = k%16,
  //                 ctile t = n/16, lane l = n%16
  // LDS index = (((s*2 + h)*4 + t)*16 + l)*16 + j
  for (int idx = threadIdx.x; idx < K * NCOLS; idx += 256) {
    const int k = idx >> 6, n = idx & 63;
    const int s = k >> 5, h = (k >> 4) & 1, j = k & 15;
    const int t = n >> 4, l = n & 15;
    ldsW[((((s << 1) + h) << 2) + t) * 256 + (l << 4) + j] = (__bf16)W[idx];
  }
  __syncthreads();

  const int lane   = threadIdx.x & 31;
  const int waveId = (blockIdx.x * (blockDim.x >> 5)) + (threadIdx.x >> 5);
  const int numRowTiles = nrows >> 4;
  if (waveId >= numRowTiles) return;           // wave-uniform: EXEC stays all-1s

  const int half = lane >> 4;                  // 0: lanes 0-15, 1: lanes 16-31
  const int l15  = lane & 15;
  const int arow = (waveId << 4) + l15;        // A-fragment row for this lane

  v8f acc[4];
#pragma unroll
  for (int t = 0; t < 4; ++t) acc[t] = (v8f){0.f,0.f,0.f,0.f,0.f,0.f,0.f,0.f};

  const int nsteps = K >> 5;
  for (int s = 0; s < nsteps; ++s) {
    // ---- A fragment: lane covers K = half*16 + [0..16) -----
    const float4* xr = reinterpret_cast<const float4*>(X + (size_t)arow * K) + (s << 3);
    float4 c0 = xr[(half << 2) + 0];
    float4 c1 = xr[(half << 2) + 1];
    float4 c2 = xr[(half << 2) + 2];
    float4 c3 = xr[(half << 2) + 3];
    v16bf a;
    a[0]=(__bf16)c0.x;  a[1]=(__bf16)c0.y;  a[2]=(__bf16)c0.z;  a[3]=(__bf16)c0.w;
    a[4]=(__bf16)c1.x;  a[5]=(__bf16)c1.y;  a[6]=(__bf16)c1.z;  a[7]=(__bf16)c1.w;
    a[8]=(__bf16)c2.x;  a[9]=(__bf16)c2.y;  a[10]=(__bf16)c2.z; a[11]=(__bf16)c2.w;
    a[12]=(__bf16)c3.x; a[13]=(__bf16)c3.y; a[14]=(__bf16)c3.z; a[15]=(__bf16)c3.w;

    // ---- B fragments from LDS: one aligned 32-byte read per lane ---------
#pragma unroll
    for (int t = 0; t < 4; ++t) {
      const v16bf b = *reinterpret_cast<const v16bf*>(
          &ldsW[((((s << 1) + half) << 2) + t) * 256 + (l15 << 4)]);
      acc[t] = __builtin_amdgcn_wmma_f32_16x16x32_bf16(
          /*neg_a=*/false, a, /*neg_b=*/false, b,
          /*c_mod=*/(short)0, acc[t], /*reuse_a=*/false, /*reuse_b=*/false);
    }
  }
  // ---- D store: VGPR r -> row = r + 8*half, col = t*16 + (lane&15) --------
#pragma unroll
  for (int t = 0; t < 4; ++t) {
#pragma unroll
    for (int r = 0; r < 8; ++r) {
      const int orow = (waveId << 4) + r + (half << 3);
      Hout[(size_t)orow * NCOLS + (t << 4) + l15] = acc[t][r];
    }
  }
}

// ---------------------------------------------------------------------------
// Per-node attention scores: aS[n,h] = <h[n,h,:], att_src[h,:]>, same for aD.
// C is a multiple of 4 -> float4 dot products.
// ---------------------------------------------------------------------------
__global__ void gat_attn_scores(const float* __restrict__ h,
                                const float* __restrict__ att_src,
                                const float* __restrict__ att_dst,
                                float* __restrict__ aS, float* __restrict__ aD,
                                int N, int Hh, int C) {
  const int tid = blockIdx.x * blockDim.x + threadIdx.x;
  if (tid >= N * Hh) return;
  const int n = tid / Hh, hh = tid - n * Hh;
  const float4* hp = reinterpret_cast<const float4*>(h + (size_t)n * Hh * C + (size_t)hh * C);
  const float4* as = reinterpret_cast<const float4*>(att_src + hh * C);
  const float4* ad = reinterpret_cast<const float4*>(att_dst + hh * C);
  float s = 0.f, d = 0.f;
  for (int c = 0; c < (C >> 2); ++c) {
    const float4 v = hp[c], a = as[c], b = ad[c];
    s += v.x * a.x + v.y * a.y + v.z * a.z + v.w * a.w;
    d += v.x * b.x + v.y * b.y + v.z * b.z + v.w * b.w;
  }
  aS[tid] = s; aD[tid] = d;
}

__global__ void fill_f32(float* __restrict__ p, float v, int n) {
  const int tid = blockIdx.x * blockDim.x + threadIdx.x;
  if (tid < n) p[tid] = v;
}

__device__ inline void atomicMaxF(float* addr, float val) {
  int* ai = reinterpret_cast<int*>(addr);
  int cur = __float_as_int(*addr);
  while (__int_as_float(cur) < val) {
    const int prev = atomicCAS(ai, cur, __float_as_int(val));
    if (prev == cur) break;
    cur = prev;
  }
}

__device__ inline void edge_sd(const int* __restrict__ ei, int e, int E,
                               int& s, int& d) {
  if (e < E) { s = ei[e]; d = ei[E + e]; }     // real edge
  else       { s = e - E; d = e - E;     }     // self-loop
}

__device__ inline float lrelu(float x) { return x > 0.f ? x : NEG_SLOPE * x; }

// ---- Pass A: per-dst running max of leaky-relu logits ---------------------
__global__ void gat_edge_max(const int* __restrict__ ei, int E, int N, int Hh,
                             const float* __restrict__ aS,
                             const float* __restrict__ aD,
                             float* __restrict__ m) {
  const int tid = blockIdx.x * blockDim.x + threadIdx.x;
  if (tid >= (E + N) * Hh) return;
  const int e = tid / Hh, hh = tid - e * Hh;
  int s, d; edge_sd(ei, e, E, s, d);
  const float l = lrelu(aS[s * Hh + hh] + aD[d * Hh + hh]);
  atomicMaxF(&m[d * Hh + hh], l);
}

// ---- Pass B: per-dst sum of exp(logit - max) ------------------------------
__global__ void gat_edge_sum(const int* __restrict__ ei, int E, int N, int Hh,
                             const float* __restrict__ aS,
                             const float* __restrict__ aD,
                             const float* __restrict__ m,
                             float* __restrict__ den) {
  const int tid = blockIdx.x * blockDim.x + threadIdx.x;
  if (tid >= (E + N) * Hh) return;
  const int e = tid / Hh, hh = tid - e * Hh;
  int s, d; edge_sd(ei, e, E, s, d);
  const float l = lrelu(aS[s * Hh + hh] + aD[d * Hh + hh]);
  atomicAdd(&den[d * Hh + hh], __expf(l - m[d * Hh + hh]));
}

// ---- Pass C: weighted scatter: acc[d,h,:] += alpha * h[s,h,:] -------------
__global__ void gat_edge_agg(const int* __restrict__ ei, int E, int N,
                             int Hh, int C,
                             const float* __restrict__ aS,
                             const float* __restrict__ aD,
                             const float* __restrict__ m,
                             const float* __restrict__ den,
                             const float* __restrict__ h,
                             float* __restrict__ acc) {
  const int tid = blockIdx.x * blockDim.x + threadIdx.x;
  if (tid >= (E + N) * Hh) return;
  const int e = tid / Hh, hh = tid - e * Hh;
  int s, d; edge_sd(ei, e, E, s, d);
  const float l = lrelu(aS[s * Hh + hh] + aD[d * Hh + hh]);
  const float alpha = __expf(l - m[d * Hh + hh]) / den[d * Hh + hh];
  const float* hp = h + (size_t)s * Hh * C + (size_t)hh * C;
  float* op = acc + (size_t)d * Hh * C + (size_t)hh * C;
  for (int c = 0; c < C; ++c) atomicAdd(&op[c], hp[c] * alpha);
}

// ---- epilogues (F == 64, n % 4 == 0 -> float4) ----------------------------
__global__ void relu_bias(const float* __restrict__ acc,
                          const float* __restrict__ b,
                          float* __restrict__ out, int n4) {
  const int tid = blockIdx.x * blockDim.x + threadIdx.x;
  if (tid >= n4) return;
  const float4 a = reinterpret_cast<const float4*>(acc)[tid];
  const float4 bb = reinterpret_cast<const float4*>(b)[tid & 15];
  float4 r;
  r.x = fmaxf(a.x + bb.x, 0.f); r.y = fmaxf(a.y + bb.y, 0.f);
  r.z = fmaxf(a.z + bb.z, 0.f); r.w = fmaxf(a.w + bb.w, 0.f);
  reinterpret_cast<float4*>(out)[tid] = r;
}

__global__ void bias_add_inplace(float* __restrict__ out,
                                 const float* __restrict__ b, int n4) {
  const int tid = blockIdx.x * blockDim.x + threadIdx.x;
  if (tid >= n4) return;
  float4 a = reinterpret_cast<float4*>(out)[tid];
  const float4 bb = reinterpret_cast<const float4*>(b)[tid & 15];
  a.x += bb.x; a.y += bb.y; a.z += bb.z; a.w += bb.w;
  reinterpret_cast<float4*>(out)[tid] = a;
}

// ---------------------------------------------------------------------------
// Host-side launch sequence
// ---------------------------------------------------------------------------
static inline int cdiv(int a, int b) { return (a + b - 1) / b; }

extern "C" void kernel_launch(void* const* d_in, const int* in_sizes, int n_in,
                              void* d_out, int out_size, void* d_ws, size_t ws_size,
                              hipStream_t stream) {
  const float* x    = (const float*)d_in[0];
  const int*   ei   = (const int*)  d_in[1];   // [2, E]
  const float* W1   = (const float*)d_in[2];   // [128, 64]
  const float* as1  = (const float*)d_in[3];   // [4, 16]
  const float* ad1  = (const float*)d_in[4];
  const float* b1   = (const float*)d_in[5];   // [64]
  const float* W2   = (const float*)d_in[6];   // [64, 64]
  const float* as2  = (const float*)d_in[7];   // [1, 64]
  const float* ad2  = (const float*)d_in[8];
  const float* b2   = (const float*)d_in[9];   // [64]

  const int Fin = 128, H1 = 4, C1 = 16, HC = 64, Fout = 64;
  const int N = in_sizes[0] / Fin;
  const int E = in_sizes[1] / 2;

  float* ws   = (float*)d_ws;
  float* bufA = ws;                            // [N*64]  h1, then relu(out1)
  float* bufB = bufA + (size_t)N * HC;         // [N*64]  acc1, then h2
  float* aS1  = bufB + (size_t)N * HC;         // [N*4]
  float* aD1  = aS1 + (size_t)N * H1;
  float* m1   = aD1 + (size_t)N * H1;
  float* dn1  = m1  + (size_t)N * H1;
  float* aS2  = dn1 + (size_t)N * H1;          // [N]
  float* aD2  = aS2 + N;
  float* m2   = aD2 + N;
  float* dn2  = m2  + N;
  float* outp = (float*)d_out;                 // [N*64] layer-2 accumulator

  const int TB = 256;
  const int gemmBlocks = cdiv((N / 16) * 32, TB);
  const int edgeTh1 = (E + N) * H1;            // layer-1 edge threads
  const int edgeTh2 = (E + N);                 // layer-2 (H=1)

  // ---------------- Layer 1 ----------------
  gat_gemm_bf16_wmma<<<gemmBlocks, TB, 0, stream>>>(x, W1, bufA, N, Fin);
  gat_attn_scores<<<cdiv(N * H1, TB), TB, 0, stream>>>(bufA, as1, ad1, aS1, aD1, N, H1, C1);
  fill_f32<<<cdiv(N * H1, TB), TB, 0, stream>>>(m1, -1e30f, N * H1);
  fill_f32<<<cdiv(N * H1, TB), TB, 0, stream>>>(dn1, 0.f, N * H1);
  fill_f32<<<cdiv(N * HC, TB), TB, 0, stream>>>(bufB, 0.f, N * HC);
  gat_edge_max<<<cdiv(edgeTh1, TB), TB, 0, stream>>>(ei, E, N, H1, aS1, aD1, m1);
  gat_edge_sum<<<cdiv(edgeTh1, TB), TB, 0, stream>>>(ei, E, N, H1, aS1, aD1, m1, dn1);
  gat_edge_agg<<<cdiv(edgeTh1, TB), TB, 0, stream>>>(ei, E, N, H1, C1,
                                                     aS1, aD1, m1, dn1, bufA, bufB);
  relu_bias<<<cdiv(N * HC / 4, TB), TB, 0, stream>>>(bufB, b1, bufA, N * HC / 4);

  // ---------------- Layer 2 ----------------
  gat_gemm_bf16_wmma<<<gemmBlocks, TB, 0, stream>>>(bufA, W2, bufB, N, HC);
  gat_attn_scores<<<cdiv(N, TB), TB, 0, stream>>>(bufB, as2, ad2, aS2, aD2, N, 1, Fout);
  fill_f32<<<cdiv(N, TB), TB, 0, stream>>>(m2, -1e30f, N);
  fill_f32<<<cdiv(N, TB), TB, 0, stream>>>(dn2, 0.f, N);
  fill_f32<<<cdiv(N * Fout, TB), TB, 0, stream>>>(outp, 0.f, N * Fout);
  gat_edge_max<<<cdiv(edgeTh2, TB), TB, 0, stream>>>(ei, E, N, 1, aS2, aD2, m2);
  gat_edge_sum<<<cdiv(edgeTh2, TB), TB, 0, stream>>>(ei, E, N, 1, aS2, aD2, m2, dn2);
  gat_edge_agg<<<cdiv(edgeTh2, TB), TB, 0, stream>>>(ei, E, N, 1, Fout,
                                                     aS2, aD2, m2, dn2, bufB, outp);
  bias_add_inplace<<<cdiv(N * Fout / 4, TB), TB, 0, stream>>>(outp, b2, N * Fout / 4);
}
